// TestMoE_52733608460613
// MI455X (gfx1250) — matrix operand order repaired and checked by reading, not compiled, not run on hardware
//
#include <hip/hip_runtime.h>

typedef __attribute__((ext_vector_type(16))) _Float16 v16h;
typedef __attribute__((ext_vector_type(8)))  float    v8f;
typedef __attribute__((ext_vector_type(4)))  int      v4i;

#define MOE_N 16384
#define MOE_D 1024
#define MOE_E 8
#define MAX_TILES 160   // sum_e ceil(count_e/128) <= 128 + 8 = 136

union Frag { uint4 q[2]; v16h h; };

// ---- CDNA5 async global->LDS copy (ASYNCcnt-tracked) -----------------------
typedef __attribute__((address_space(1))) v4i g_v4i;   // global int4
typedef __attribute__((address_space(3))) v4i l_v4i;   // LDS int4

__device__ __forceinline__ void async_b128(const _Float16* g, _Float16* l) {
#if __has_builtin(__builtin_amdgcn_global_load_async_to_lds_b128)
  __builtin_amdgcn_global_load_async_to_lds_b128((g_v4i*)g, (l_v4i*)l, 0, 0);
#else
  const unsigned loff = (unsigned)(uintptr_t)l;  // LDS byte offset (low 32b)
  asm volatile("global_load_async_to_lds_b128 %0, %1, off"
               :: "v"(loff), "v"(g) : "memory");
#endif
}

__device__ __forceinline__ void wait_async0() {
#if __has_builtin(__builtin_amdgcn_s_wait_asynccnt)
  __builtin_amdgcn_s_wait_asynccnt(0);
#else
  asm volatile("s_wait_asynccnt 0x0" ::: "memory");
#endif
}

__device__ __forceinline__ uint2 pack4h(float a, float b, float c, float d) {
  union { _Float16 h[4]; uint2 u; } t;
  t.h[0] = (_Float16)a; t.h[1] = (_Float16)b;
  t.h[2] = (_Float16)c; t.h[3] = (_Float16)d;
  return t.u;
}

// ---------------- expert weight fp32 -> fp16 (layout preserved [E][out][in]) --
__global__ __launch_bounds__(256) void moe_cvt_w(const float* __restrict__ w,
                                                 _Float16* __restrict__ wf) {
  size_t i = ((size_t)blockIdx.x * 256 + threadIdx.x) * 4;
  float4 v = *(const float4*)(w + i);
  *(uint2*)(wf + i) = pack4h(v.x, v.y, v.z, v.w);
}

// ---------------- router: one wave per token; fused x -> fp16 convert --------
__global__ __launch_bounds__(256) void moe_router(
    const float* __restrict__ x, const float* __restrict__ rw,
    const float* __restrict__ rb, int* __restrict__ top,
    int* __restrict__ counts, _Float16* __restrict__ xf) {
  const int token = (int)((blockIdx.x * 256u + threadIdx.x) >> 5);
  const int lane  = threadIdx.x & 31;
  const float4* xr = (const float4*)(x + (size_t)token * MOE_D);
  _Float16* xo = xf + (size_t)token * MOE_D;
  float acc[MOE_E];
#pragma unroll
  for (int e = 0; e < MOE_E; ++e) acc[e] = 0.f;
#pragma unroll
  for (int it = 0; it < 8; ++it) {
    const int idx = lane + it * 32;         // float4 index within the row
    const float4 xv = xr[idx];
    *(uint2*)(xo + idx * 4) = pack4h(xv.x, xv.y, xv.z, xv.w);
#pragma unroll
    for (int e = 0; e < MOE_E; ++e) {
      const float4 wv = ((const float4*)(rw + e * MOE_D))[idx];
      acc[e] += xv.x * wv.x + xv.y * wv.y + xv.z * wv.z + xv.w * wv.w;
    }
  }
#pragma unroll
  for (int e = 0; e < MOE_E; ++e) {
    float v = acc[e];
#pragma unroll
    for (int off = 16; off > 0; off >>= 1) v += __shfl_xor(v, off, 32);
    acc[e] = v;
  }
  if (lane == 0) {
    float best = acc[0] + rb[0]; int bi = 0;
#pragma unroll
    for (int e = 1; e < MOE_E; ++e) {
      float v = acc[e] + rb[e];
      if (v > best) { best = v; bi = e; }   // first-max wins, matches argmax
    }
    top[token] = bi;
    atomicAdd(&counts[bi], 1);
  }
}

// ---------------- offsets + tile map (trivial, E=8) --------------------------
__global__ void moe_build(const int* __restrict__ counts, int* __restrict__ offsets,
                          int* __restrict__ cursor, int* __restrict__ tileE,
                          int* __restrict__ tileS) {
  if (threadIdx.x == 0 && blockIdx.x == 0) {
    int off = 0, t = 0;
    for (int e = 0; e < MOE_E; ++e) {
      offsets[e] = off; cursor[e] = off;
      for (int st = 0; st < counts[e]; st += 128) { tileE[t] = e; tileS[t] = st; ++t; }
      off += counts[e];
    }
    for (; t < MAX_TILES; ++t) tileE[t] = -1;
  }
}

// ---------------- scatter: token -> per-expert index list --------------------
__global__ __launch_bounds__(256) void moe_scatter(const int* __restrict__ top,
                                                   int* __restrict__ cursor,
                                                   int* __restrict__ idx_list) {
  const int n = blockIdx.x * 256 + threadIdx.x;
  const int e = top[n];
  const int p = atomicAdd(&cursor[e], 1);
  idx_list[p] = n;
}

// ---------------- grouped GEMM: C[tile] = Xgathered * W[e]^T + b[e] ----------
// Block tile: 128 (tokens) x 128 (out cols), K stepped by 32 (one WMMA depth).
// 8 waves, each wave owns a 64x32 sub-tile = 4x2 WMMA accumulators.
// Global->LDS staging uses CDNA5 async DMA (ASYNCcnt) with ping-pong buffers.
__global__ __launch_bounds__(256) void moe_gemm(
    const _Float16* __restrict__ xf, const _Float16* __restrict__ wf,
    const float* __restrict__ eb, const int* __restrict__ counts,
    const int* __restrict__ offsets, const int* __restrict__ tileE,
    const int* __restrict__ tileS, const int* __restrict__ idx_list,
    float* __restrict__ out) {
  __shared__ __align__(16) _Float16 Alds[2][128 * 40];  // padded stride 40 halves
  __shared__ __align__(16) _Float16 Blds[2][128 * 40];

  const int e = tileE[blockIdx.x];
  if (e < 0) return;
  const int st = tileS[blockIdx.x];
  const int s  = offsets[e] + st;                 // gathered-row start
  const int rowsValid = min(128, counts[e] - st); // >= 1 by construction
  const int n0 = blockIdx.y * 128;                // output-column tile start

  // --- staging: thread t owns row t>>1, halves [(t&1)*16 .. +15] ------------
  const int sr = threadIdx.x >> 1;
  const int sc = (threadIdx.x & 1) << 4;          // 0 or 16 halves
  const bool aV = sr < rowsValid;
  // invalid tail rows DMA a *valid* token row (finite garbage, never stored)
  const int tokRow = idx_list[s + (aV ? sr : 0)];
  const _Float16* gA = xf + (size_t)tokRow * MOE_D + sc;
  const _Float16* gB = wf + ((size_t)e << 20) + (size_t)(n0 + sr) * MOE_D + sc;
  _Float16* lA[2] = { &Alds[0][sr * 40 + sc], &Alds[1][sr * 40 + sc] };
  _Float16* lB[2] = { &Blds[0][sr * 40 + sc], &Blds[1][sr * 40 + sc] };

  // prologue: DMA buffer 0
  async_b128(gA,     lA[0]);  async_b128(gA + 8, lA[0] + 8);
  async_b128(gB,     lB[0]);  async_b128(gB + 8, lB[0] + 8);
  wait_async0();
  __syncthreads();

  const int lane = threadIdx.x & 31;
  const int waveId = threadIdx.x >> 5;
  const int wm = waveId & 1;    // 2 waves over M (64 rows each)
  const int wn = waveId >> 1;   // 4 waves over N (32 cols each)
  const int lm = lane & 15;
  const int lh = lane >> 4;     // lane group 0/1

  v8f c[4][2] = {};

  for (int kt = 0; kt < MOE_D / 32; ++kt) {
    const int buf = kt & 1;
    if (kt < MOE_D / 32 - 1) {               // DMA next K-slab into other buffer
      const int k = (kt + 1) * 32;
      const int nb = buf ^ 1;
      async_b128(gA + k,     lA[nb]);  async_b128(gA + k + 8, lA[nb] + 8);
      async_b128(gB + k,     lB[nb]);  async_b128(gB + k + 8, lB[nb] + 8);
    }

    // A fragments: lane group 0 -> K{0..7,16..23}; group 1 -> K{8..15,24..31}
    Frag a[4], b[2];
#pragma unroll
    for (int mi = 0; mi < 4; ++mi) {
      const _Float16* p = &Alds[buf][(wm * 64 + mi * 16 + lm) * 40 + lh * 8];
      a[mi].q[0] = *(const uint4*)(p);
      a[mi].q[1] = *(const uint4*)(p + 16);
    }
    // B fragments: lane group 0 -> K{0..7,16..23}; group 1 -> K{8..15,24..31}
#pragma unroll
    for (int ni = 0; ni < 2; ++ni) {
      const _Float16* p = &Blds[buf][(wn * 32 + ni * 16 + lm) * 40 + lh * 8];
      b[ni].q[0] = *(const uint4*)(p);
      b[ni].q[1] = *(const uint4*)(p + 16);
    }
#pragma unroll
    for (int mi = 0; mi < 4; ++mi)
#pragma unroll
      for (int ni = 0; ni < 2; ++ni)
        c[mi][ni] = __builtin_amdgcn_wmma_f32_16x16x32_f16(
            false, a[mi].h, false, b[ni].h, (short)0, c[mi][ni], false, false);

    wait_async0();        // own DMAs into next buffer are complete
    __syncthreads();      // everyone's DMAs visible before next iteration
  }

  // epilogue: bias + indirect row store (C: VGPR j -> M=j+8*lh, N=lm)
  const int nbase = n0 + wn * 32 + lm;
  const float b0 = eb[e * MOE_D + nbase];
  const float b1 = eb[e * MOE_D + nbase + 16];
#pragma unroll
  for (int mi = 0; mi < 4; ++mi) {
#pragma unroll
    for (int j = 0; j < 8; ++j) {
      const int rl = wm * 64 + mi * 16 + lh * 8 + j;
      if (rl < rowsValid) {
        const int tok = idx_list[s + rl];
        float* orow = out + (size_t)tok * MOE_D + nbase;
        __builtin_nontemporal_store(c[mi][0][j] + b0, orow);
        __builtin_nontemporal_store(c[mi][1][j] + b1, orow + 16);
      }
    }
  }
}

extern "C" void kernel_launch(void* const* d_in, const int* in_sizes, int n_in,
                              void* d_out, int out_size, void* d_ws, size_t ws_size,
                              hipStream_t stream) {
  (void)in_sizes; (void)n_in; (void)out_size; (void)ws_size;
  const float* x  = (const float*)d_in[0];
  const float* rw = (const float*)d_in[1];
  const float* rb = (const float*)d_in[2];
  const float* ew = (const float*)d_in[3];
  const float* eb = (const float*)d_in[4];
  float* out = (float*)d_out;

  // workspace layout (needs ~51 MB)
  uint8_t* w = (uint8_t*)d_ws;
  int* counts   = (int*)w;            // 8
  int* offsets  = counts + 8;         // 8
  int* cursor   = counts + 16;        // 8
  int* tileE    = counts + 24;        // MAX_TILES
  int* tileS    = tileE + MAX_TILES;  // MAX_TILES
  int* top      = (int*)(w + 4096);           // N ints
  int* idx_list = top + MOE_N;                // N ints
  _Float16* xf  = (_Float16*)(w + (1u  << 20));   // 32 MB
  _Float16* wf  = (_Float16*)(w + (35u << 20));   // 16 MB

  (void)hipMemsetAsync(counts, 0, MOE_E * sizeof(int), stream);
  moe_cvt_w<<<(MOE_E * MOE_D * MOE_D) / (256 * 4), 256, 0, stream>>>(ew, wf);
  moe_router<<<(MOE_N * 32) / 256, 256, 0, stream>>>(x, rw, rb, top, counts, xf);
  moe_build<<<1, 32, 0, stream>>>(counts, offsets, cursor, tileE, tileS);
  moe_scatter<<<MOE_N / 256, 256, 0, stream>>>(top, cursor, idx_list);
  moe_gemm<<<dim3(MAX_TILES, MOE_D / 128), 256, 0, stream>>>(
      xf, wf, eb, counts, offsets, tileE, tileS, idx_list, out);
}
